// TPAsyncDense_9294309228963
// MI455X (gfx1250) — compile-verified, matrix operand order checked by reading, not run on hardware
//
#include <hip/hip_runtime.h>

typedef __attribute__((ext_vector_type(16))) __bf16 v16bf;
typedef __attribute__((ext_vector_type(8)))  float  v8f;

#define TPX   8
#define MTOT  16384             // B*S rows per device
#define DKF   256
#define DOF   512

#define BM 128
#define BN 128
#define BK 32
#define NCHUNK 64               // 8 ring steps * 8 k-chunks
#define LDA 40                  // bf16 elems per LDS row (32 + 8 pad), 80B = 5*16B
#define LDB 40

#define OFF_AH 0
#define OFF_AL 10240
#define OFF_BH 20480
#define OFF_BL 30720

struct Frag { union { v16bf v; uint4 q[2]; }; };

// {b[31:16], a[31:16]} in one v_perm_b32
__device__ __forceinline__ unsigned pack_hi16(unsigned a, unsigned b) {
    return __builtin_amdgcn_perm(b, a, 0x07060302u);
}

__global__ __launch_bounds__(256)
void tp_async_dense_bf16x3(const float* __restrict__ x,
                           const float* __restrict__ W,
                           const float* __restrict__ b,
                           float* __restrict__ y) {
    __shared__ __align__(16) unsigned char smem[65536];
    unsigned short* Ah = (unsigned short*)(smem + OFF_AH);
    unsigned short* Al = (unsigned short*)(smem + OFF_AL);
    unsigned short* Bh = (unsigned short*)(smem + OFF_BH);
    unsigned short* Bl = (unsigned short*)(smem + OFF_BL);
    float* Cout = (float*)smem;   // epilogue reuses all 64KB

    const int tid   = threadIdx.x;
    const int lane  = tid & 31;
    const int wave  = tid >> 5;          // 8 waves
    const int wm    = wave >> 1;         // 0..3  (32-row strip)
    const int wn    = wave & 1;          // 0..1  (64-col strip)
    const int l16   = lane & 15;
    const int khalf = lane >> 4;         // 0|1
    const int k0    = khalf * 8;         // bf16 K-run start per ISA 16-bit layout

    // tile decode: 8 devices * 128 m-tiles * 4 n-tiles = 4096 blocks
    const int t  = blockIdx.x;
    const int d  = t >> 9;
    const int tm = (t & 511) >> 2;
    const int tn = t & 3;
    const size_t m0 = (size_t)tm * BM;
    const int    n0 = tn * BN;

    // per-thread fixed staging coordinates
    const int rA = tid >> 3;             // A row base (0..31), +32 per it
    const int cA = (tid & 7) << 2;       // A col (0..28)
    const int kB = tid >> 5;             // B k base (0..7), +8 per it
    const int nB = (tid & 31) << 2;      // B n (0..124)

    v8f acc[2][4];
#pragma unroll
    for (int mt = 0; mt < 2; ++mt)
#pragma unroll
        for (int nt = 0; nt < 4; ++nt)
            acc[mt][nt] = (v8f){0.f,0.f,0.f,0.f,0.f,0.f,0.f,0.f};

    float4 ga[4], gb[4];

    auto loadGlobal = [&](int c) {
        const int i  = c >> 3;
        const int kc = (c & 7) << 5;
        const int s  = (d - i) & 7;
        const float* baseA = x + ((size_t)s * MTOT + m0) * DKF + kc;
        const float* baseB = W + (((size_t)d * TPX + i) * DKF + kc) * DOF + n0;
#pragma unroll
        for (int it = 0; it < 4; ++it) {
            ga[it] = *(const float4*)(baseA + (size_t)(rA + it * 32) * DKF + cA);
            gb[it] = *(const float4*)(baseB + (size_t)(kB + it * 8) * DOF + nB);
        }
    };

    auto storeStage = [&]() {
#pragma unroll
        for (int it = 0; it < 4; ++it) {
            // ---- A: 4 consecutive k of one row -> packed uint2 hi/lo ----
            float4 f = ga[it];
            unsigned a0 = __float_as_uint(f.x), a1 = __float_as_uint(f.y);
            unsigned a2 = __float_as_uint(f.z), a3 = __float_as_uint(f.w);
            unsigned r0 = __float_as_uint(f.x - __uint_as_float(a0 & 0xffff0000u));
            unsigned r1 = __float_as_uint(f.y - __uint_as_float(a1 & 0xffff0000u));
            unsigned r2 = __float_as_uint(f.z - __uint_as_float(a2 & 0xffff0000u));
            unsigned r3 = __float_as_uint(f.w - __uint_as_float(a3 & 0xffff0000u));
            uint2 ph = make_uint2(pack_hi16(a0, a1), pack_hi16(a2, a3));
            uint2 pl = make_uint2(pack_hi16(r0, r1), pack_hi16(r2, r3));
            int r = rA + it * 32;
            *(uint2*)(Ah + r * LDA + cA) = ph;
            *(uint2*)(Al + r * LDA + cA) = pl;

            // ---- B: 4 consecutive n of one k -> transposed [n][k] scatter ----
            float4 g = gb[it];
            int k = kB + it * 8;
            unsigned b0 = __float_as_uint(g.x), b1 = __float_as_uint(g.y);
            unsigned b2 = __float_as_uint(g.z), b3 = __float_as_uint(g.w);
            unsigned s0 = __float_as_uint(g.x - __uint_as_float(b0 & 0xffff0000u));
            unsigned s1 = __float_as_uint(g.y - __uint_as_float(b1 & 0xffff0000u));
            unsigned s2 = __float_as_uint(g.z - __uint_as_float(b2 & 0xffff0000u));
            unsigned s3 = __float_as_uint(g.w - __uint_as_float(b3 & 0xffff0000u));
            Bh[(nB+0)*LDB + k] = (unsigned short)(b0 >> 16);
            Bh[(nB+1)*LDB + k] = (unsigned short)(b1 >> 16);
            Bh[(nB+2)*LDB + k] = (unsigned short)(b2 >> 16);
            Bh[(nB+3)*LDB + k] = (unsigned short)(b3 >> 16);
            Bl[(nB+0)*LDB + k] = (unsigned short)(s0 >> 16);
            Bl[(nB+1)*LDB + k] = (unsigned short)(s1 >> 16);
            Bl[(nB+2)*LDB + k] = (unsigned short)(s2 >> 16);
            Bl[(nB+3)*LDB + k] = (unsigned short)(s3 >> 16);
        }
    };

    auto computeChunk = [&]() {
        Frag ah[2], al[2], bh[4], bl[4];
#pragma unroll
        for (int mt = 0; mt < 2; ++mt) {
            const unsigned short* pa = Ah + (size_t)(wm*32 + mt*16 + l16) * LDA + k0;
            const unsigned short* pl = Al + (size_t)(wm*32 + mt*16 + l16) * LDA + k0;
            ah[mt].q[0] = *(const uint4*)pa;  ah[mt].q[1] = *(const uint4*)(pa + 16);
            al[mt].q[0] = *(const uint4*)pl;  al[mt].q[1] = *(const uint4*)(pl + 16);
        }
#pragma unroll
        for (int nt = 0; nt < 4; ++nt) {
            const unsigned short* pb = Bh + (size_t)(wn*64 + nt*16 + l16) * LDB + k0;
            const unsigned short* pl = Bl + (size_t)(wn*64 + nt*16 + l16) * LDB + k0;
            bh[nt].q[0] = *(const uint4*)pb;  bh[nt].q[1] = *(const uint4*)(pb + 16);
            bl[nt].q[0] = *(const uint4*)pl;  bl[nt].q[1] = *(const uint4*)(pl + 16);
        }
        // bf16x3 emulated-f32 MACs: ah*bh + ah*bl + al*bh
#pragma unroll
        for (int mt = 0; mt < 2; ++mt)
#pragma unroll
            for (int nt = 0; nt < 4; ++nt) {
                acc[mt][nt] = __builtin_amdgcn_wmma_f32_16x16x32_bf16(
                    false, ah[mt].v, false, bh[nt].v, (short)0, acc[mt][nt], false, false);
                acc[mt][nt] = __builtin_amdgcn_wmma_f32_16x16x32_bf16(
                    false, ah[mt].v, false, bl[nt].v, (short)0, acc[mt][nt], false, false);
                acc[mt][nt] = __builtin_amdgcn_wmma_f32_16x16x32_bf16(
                    false, al[mt].v, false, bh[nt].v, (short)0, acc[mt][nt], false, false);
            }
    };

    // ---- software pipeline: prefetch chunk c+1 (regs) under chunk c compute ----
    loadGlobal(0);
    for (int c = 0; c < NCHUNK; ++c) {
        __syncthreads();                    // all waves done reading LDS (prev chunk)
        storeStage();                       // convert + ds_store chunk c
        if (c + 1 < NCHUNK) loadGlobal(c + 1);   // issue HBM/L2 loads; waited next iter
        __syncthreads();
        computeChunk();                     // 24 wmma, hides global latency
    }

    // ---- epilogue: stage C tile in LDS, then coalesced b128 stores + bias ----
    __syncthreads();
#pragma unroll
    for (int mt = 0; mt < 2; ++mt)
#pragma unroll
        for (int nt = 0; nt < 4; ++nt) {
            int col = wn*64 + nt*16 + l16;
#pragma unroll
            for (int r = 0; r < 8; ++r) {
                int row = wm*32 + mt*16 + khalf*8 + r;    // C layout: vgpr r -> M=r / M=8+r
                Cout[row * BN + col] = acc[mt][nt][r];
            }
        }
    __syncthreads();

    const float* bd = b + d * DOF;
#pragma unroll
    for (int it = 0; it < 16; ++it) {
        int idx = tid + it * 256;                          // 4096 float4 = 128x128
        int row = idx >> 5;
        int c4  = (idx & 31) << 2;
        float4 v = *(const float4*)(Cout + row * BN + c4);
        int gn = n0 + c4;
        v.x += bd[gn]; v.y += bd[gn+1]; v.z += bd[gn+2]; v.w += bd[gn+3];
        *(float4*)(y + ((size_t)d * MTOT + m0 + row) * DOF + gn) = v;
    }
}

extern "C" void kernel_launch(void* const* d_in, const int* in_sizes, int n_in,
                              void* d_out, int out_size, void* d_ws, size_t ws_size,
                              hipStream_t stream) {
    const float* x = (const float*)d_in[0];   // (8, 8, 2048, 256) f32
    const float* W = (const float*)d_in[1];   // (8, 8, 256, 512)  f32
    const float* b = (const float*)d_in[2];   // (8, 512)          f32
    float* y = (float*)d_out;                 // (8, 8, 2048, 512) f32

    dim3 grid(4096);   // 8 devices * 128 m-tiles * 4 n-tiles
    dim3 block(256);   // 8 wave32
    hipLaunchKernelGGL(tp_async_dense_bf16x3, grid, block, 0, stream, x, W, b, y);
}
